// GraphBP_13675175870753
// MI455X (gfx1250) — matrix operand order repaired and checked by reading, not compiled, not run on hardware
//
#include <hip/hip_runtime.h>
#include <hip/hip_bf16.h>

// ---------------------------------------------------------------------------
// GraphBP forward for MI455X (gfx1250, wave32, WMMA).
// Heavy GEMMs run as f16 WMMA 16x16x32 with f32 accumulate; intermediates are
// stored f16 so the [E,256] edge tensors (~168MB) are L2-resident (192MB L2).
// GEMM blocks compute 128x32 tiles, K-step 64: each wave owns 2 M-sub-tiles
// and issues 8 WMMAs per K-step. A tiles are staged with CDNA5 async
// global->LDS DMA (GLOBAL_LOAD_ASYNC_TO_LDS_B128 / ASYNCcnt), double-buffered
// so tile k+1's DMA overlaps tile k's WMMAs with a single barrier per step.
// ---------------------------------------------------------------------------

typedef _Float16 f16;
typedef __attribute__((ext_vector_type(16))) _Float16 v16h;
typedef __attribute__((ext_vector_type(8)))  float    v8f;

#define NB    512
#define NATOM 40
#define NN    (NB*NATOM)      // 20480 nodes
#define DEG   16
#define NE    (NN*DEG)        // 327680 edges
#define HH    256
#define GG    50
#define LL    3
#define KK    6
#define AA    10
#define RR    6
#define BZC   8
#define NLIG  (NN/2)
#define CUT   10.0f
#define PIF   3.14159265358979f

#define ACT_NONE 0
#define ACT_SSP  1
#define ACT_RELU 2
#define ACT_TANH 3

__device__ __constant__ float SPHZ[3][6] = {
  {3.14159265f, 6.28318531f, 9.42477796f, 12.56637061f, 15.70796327f, 18.84955592f},
  {4.49340946f, 7.72525184f, 10.90412166f, 14.06619391f, 17.22075527f, 20.37130296f},
  {5.76345920f, 9.09501133f, 12.32294097f, 15.51460301f, 18.68903635f, 21.85387222f}};

union Frag16 { v16h v; unsigned u[8]; };

__device__ __forceinline__ float actf(int act, float x) {
  if (act == ACT_SSP) {                       // shifted softplus
    float ax = fabsf(x);
    return fmaxf(x, 0.f) + log1pf(__expf(-ax)) - 0.6931472f;
  }
  if (act == ACT_RELU) return fmaxf(x, 0.f);
  if (act == ACT_TANH) return tanhf(x);
  return x;
}

// ---------------------------------------------------------------------------
// WMMA GEMM:  D = [resid +] [rowscale *] act(A[M,K] @ Bw[K,Nld] + bias)
// A rows optionally gathered through gidx (hoisted out of the K loop).
// 128 threads = 4 waves. Block tile 128 x (16*NT), K-step 64.
// Software pipeline (one barrier per K-step):
//   1) s_wait_asynccnt 0       -> tile k's A DMA (issued last step) landed
//   2) __syncthreads()         -> tile k's B visible; (k+1)&1 buffers free
//   3) compute tile k          -> 8 WMMAs (2 k-subs x 2 M-tiles x NT)
//   4) async-DMA A(k+1) + stage B(k+1) into the opposite-parity buffers
// Requires: M%128==0, K%64==0, Nld%(16*NT)==0.
// ---------------------------------------------------------------------------
template<int NT>
__global__ __launch_bounds__(128)
void gemm_wmma(const f16* __restrict__ A, const f16* __restrict__ Bw,
               const float* __restrict__ bias, const float* __restrict__ rowscale,
               const float* __restrict__ resid, f16* __restrict__ out16,
               float* __restrict__ out32, const int* __restrict__ gidx,
               int M, int K, int Nld, int act)
{
  __shared__ __align__(16) f16 As[2][128 * 64];
  __shared__ __align__(16) f16 Bs[2][NT * 16 * 64];

  const int tid  = threadIdx.x;
  const int wave = tid >> 5;
  const int lane = tid & 31;
  const int lm   = lane & 15;   // row-in-tile (A) / column (B,C)
  const int hi   = lane >> 4;   // lane-group select
  const int m_blk = blockIdx.x * 128;
  const int n0    = blockIdx.y * (16 * NT);

  // A staging slot: thread -> full 64-half row, fixed across the K loop.
  const int ar   = m_blk + tid;
  const int arow = gidx ? gidx[ar] : ar;   // hoisted gather
  const f16* aptr = A + (size_t)arow * K;
  // LDS byte addresses of this thread's A row in each buffer
  // (generic ptr low 32 bits = LDS offset)
  const unsigned lds_a[2] = {
    (unsigned)(unsigned long long)&As[0][tid * 64],
    (unsigned)(unsigned long long)&As[1][tid * 64] };

  // B staging slot: thread -> (kk, n-strip), fixed across the K loop.
  const int bkk  = tid & 63;               // k within 64
  const int bnst = (tid >> 6) * (NT * 8);  // n start within block tile
  const f16* bptr = Bw + (size_t)bkk * Nld + n0 + bnst;

  // async-DMA one 128B A row slice (8 x 16B); ISA adds the instruction
  // offset to BOTH the LDS and global addresses.
  auto issueA = [&](int k0, unsigned lds) {
    const f16* g = aptr + k0;
    asm volatile(
      "global_load_async_to_lds_b128 %0, %1, off\n\t"
      "global_load_async_to_lds_b128 %0, %1, off offset:16\n\t"
      "global_load_async_to_lds_b128 %0, %1, off offset:32\n\t"
      "global_load_async_to_lds_b128 %0, %1, off offset:48\n\t"
      "global_load_async_to_lds_b128 %0, %1, off offset:64\n\t"
      "global_load_async_to_lds_b128 %0, %1, off offset:80\n\t"
      "global_load_async_to_lds_b128 %0, %1, off offset:96\n\t"
      "global_load_async_to_lds_b128 %0, %1, off offset:112"
      :: "v"(lds), "v"(g) : "memory");
  };
  // stage B transposed: Bs[buf][n*64 + kk] = Bw[(k0+kk)*Nld + n0+n]
  auto stageB = [&](int k0, int buf) {
    const f16* src = bptr + (size_t)k0 * Nld;
    union { uint4 q[NT]; f16 h[NT * 8]; } bv;
#pragma unroll
    for (int q = 0; q < NT; ++q) bv.q[q] = ((const uint4*)src)[q];
#pragma unroll
    for (int c = 0; c < NT * 8; ++c)
      Bs[buf][(bnst + c) * 64 + bkk] = bv.h[c];
  };

  v8f acc[2][NT] = {};

  // prologue: bring tile 0 in flight
  issueA(0, lds_a[0]);
  stageB(0, 0);

  const int nk = K >> 6;
  for (int kt = 0; kt < nk; ++kt) {
    const int cur = kt & 1;
    // tile kt's A DMA has landed; publish tile kt's B; recycle old buffers
    asm volatile("s_wait_asynccnt 0" ::: "memory");
    __syncthreads();

    const f16* AsC = As[cur];
    const f16* BsC = Bs[cur];
#pragma unroll
    for (int s = 0; s < 2; ++s) {
      // A fragments per ISA 16-bit 16x32 layout (k-sub s), one per M-tile
      Frag16 fa[2];
#pragma unroll
      for (int mt = 0; mt < 2; ++mt) {
        const unsigned* As32 =
            (const unsigned*)&AsC[(wave * 32 + mt * 16 + lm) * 64];
#pragma unroll
        for (int r = 0; r < 8; ++r) {
          int kbA = ((r < 4) ? (2 * r) : (2 * (r - 4) + 16)) + hi * 8 + s * 32;
          fa[mt].u[r] = As32[kbA >> 1];
        }
      }
#pragma unroll
      for (int nt = 0; nt < NT; ++nt) {
        Frag16 fb;
        const unsigned* Bs32 = (const unsigned*)&BsC[(nt * 16 + lm) * 64];
#pragma unroll
        for (int r = 0; r < 8; ++r) {
          int kbB = ((r < 4) ? (2 * r) : (2 * (r - 4) + 16)) + hi * 8 + s * 32;
          fb.u[r] = Bs32[kbB >> 1];
        }
        acc[0][nt] = __builtin_amdgcn_wmma_f32_16x16x32_f16(
            false, fa[0].v, false, fb.v, (short)0, acc[0][nt], false, false);
        acc[1][nt] = __builtin_amdgcn_wmma_f32_16x16x32_f16(
            false, fa[1].v, false, fb.v, (short)0, acc[1][nt], false, false);
      }
    }

    // bring tile kt+1 in flight (opposite parity buffers are free)
    if (kt + 1 < nk) {
      issueA((kt + 1) << 6, lds_a[cur ^ 1]);
      stageB((kt + 1) << 6, cur ^ 1);
      if (kt + 2 < nk) __builtin_prefetch(aptr + ((kt + 2) << 6), 0, 1);
    }
  }

  // epilogue: acc[mt][nt] VGPR r -> row m0 + mt*16 + r + 8*hi, col n0+nt*16+lm
  const int m0 = m_blk + wave * 32;
#pragma unroll
  for (int mt = 0; mt < 2; ++mt) {
#pragma unroll
    for (int nt = 0; nt < NT; ++nt) {
      const int nc = n0 + nt * 16 + lm;
      float bn = bias ? bias[nc] : 0.f;
#pragma unroll
      for (int r = 0; r < 8; ++r) {
        int m = m0 + mt * 16 + r + hi * 8;
        float x = acc[mt][nt][r] + bn;
        x = actf(act, x);
        if (rowscale) x *= rowscale[m];
        size_t o = (size_t)m * Nld + nc;
        if (resid) x += resid[o];
        if (out32) out32[o] = x;
        if (out16) out16[o] = (f16)x;
      }
    }
  }
}

// ---------------------------------------------------------------------------
// elementwise / glue kernels
// ---------------------------------------------------------------------------
__global__ void k_prep_edges(const float* __restrict__ pos, const int* __restrict__ esrc,
                             const int* __restrict__ edst, f16* __restrict__ rbf16,
                             float* __restrict__ Cw)
{
  long t = (long)blockIdx.x * 256 + threadIdx.x;
  long e = t >> 6;
  int  g = (int)(t & 63);
  if (e >= NE) return;
  int i = esrc[e], j = edst[e];
  float dx = pos[i*3+0] - pos[j*3+0];
  float dy = pos[i*3+1] - pos[j*3+1];
  float dz = pos[i*3+2] - pos[j*3+2];
  float d = sqrtf(dx*dx + dy*dy + dz*dz + 1e-12f);
  float val = 0.f;
  if (g < GG) {
    float step = CUT / (float)(GG - 1);
    float coeff = -0.5f / (step * step);
    float dd = d - (float)g * step;
    val = __expf(coeff * dd * dd);
  }
  rbf16[e * 64 + g] = (f16)val;
  if (g == 0) Cw[e] = 0.5f * (__cosf(d * PIF / CUT) + 1.f);
}

__global__ void k_cvt_pad(const float* __restrict__ src, f16* __restrict__ dst,
                          int Rs, int Cs, int Rd, int Cd, long total)
{
  long idx = (long)blockIdx.x * 256 + threadIdx.x;
  if (idx >= total) return;
  long per = (long)Rd * Cd;
  int b = (int)(idx / per);
  long rem = idx - (long)b * per;
  int r = (int)(rem / Cd), c = (int)(rem % Cd);
  float v = (r < Rs && c < Cs) ? src[((long)b * Rs + r) * Cs + c] : 0.f;
  dst[idx] = (f16)v;
}

__global__ void k_pad32(const float* __restrict__ src, float* __restrict__ dst,
                        int Cs, int Cd, int nb)
{
  int idx = blockIdx.x * 256 + threadIdx.x;
  if (idx >= nb * Cd) return;
  int b = idx / Cd, c = idx % Cd;
  dst[idx] = (c < Cs) ? src[b * Cs + c] : 0.f;
}

__global__ void k_embed(const float* __restrict__ emb, const int* __restrict__ atype,
                        float* __restrict__ v32, f16* __restrict__ v16)
{
  long idx = (long)blockIdx.x * 256 + threadIdx.x;
  if (idx >= (long)NN * HH) return;
  int n = (int)(idx >> 8), h = (int)(idx & 255);
  float v = emb[atype[n] * HH + h];
  v32[idx] = v; v16[idx] = (f16)v;
}

// agg[i,:] = sum_{e<16} vl16[dst[i*16+e],:] * W16[i*16+e,:]
__global__ void k_edge_agg(const f16* __restrict__ vl16, const f16* __restrict__ W16,
                           const int* __restrict__ edst, f16* __restrict__ agg16)
{
  int i = blockIdx.x, t = threadIdx.x;
  float acc = 0.f;
  const f16* wrow = W16 + (size_t)i * DEG * HH;
#pragma unroll
  for (int e = 0; e < DEG; ++e) {
    int j = edst[i * DEG + e];
    acc += (float)vl16[(size_t)j * HH + t] * (float)wrow[e * HH + t];
  }
  agg16[(size_t)i * HH + t] = (f16)acc;
}

__global__ void k_node_emb(const float* __restrict__ v32, const float* __restrict__ emb,
                           const int* __restrict__ nat, const int* __restrict__ batch,
                           float* __restrict__ ne32, f16* __restrict__ ne16)
{
  long idx = (long)blockIdx.x * 256 + threadIdx.x;
  if (idx >= (long)NN * HH) return;
  int n = (int)(idx >> 8), h = (int)(idx & 255);
  float v = v32[idx] * emb[nat[batch[n]] * HH + h];
  ne32[idx] = v; ne16[idx] = (f16)v;
}

__global__ void k_dist_basis_mul(const float* __restrict__ nd, const int* __restrict__ batch,
                                 const float* __restrict__ w1, const float* __restrict__ w2,
                                 float* __restrict__ ne32, f16* __restrict__ ne16)
{
  int n = blockIdx.x, h = threadIdx.x;
  float d = nd[batch[n]];
  float x = fmaxf(d / CUT, 1e-6f);
  float x2 = x * x, x5 = x2 * x2 * x;
  float env = 1.f / x - 28.f * x5 + 48.f * x5 * x - 21.f * x5 * x2;
  float tb[BZC];
#pragma unroll
  for (int z = 0; z < BZC; ++z) tb[z] = 0.f;
#pragma unroll
  for (int r = 0; r < RR; ++r) {
    float rb = env * __sinf((float)(r + 1) * PIF * x);
#pragma unroll
    for (int z = 0; z < BZC; ++z) tb[z] += rb * w1[r * BZC + z];
  }
  float fac = 0.f;
#pragma unroll
  for (int z = 0; z < BZC; ++z) fac += tb[z] * w2[z * HH + h];
  size_t o = (size_t)n * HH + h;
  float v = ne32[o] * fac;
  ne32[o] = v; ne16[o] = (f16)v;
}

__global__ void k_sph_basis_mul(const float* __restrict__ nd, const float* __restrict__ na,
                                const int* __restrict__ batch,
                                const float* __restrict__ w1, const float* __restrict__ w2,
                                float* __restrict__ ne32, f16* __restrict__ ne16)
{
  int n = blockIdx.x, h = threadIdx.x;
  int b = batch[n];
  float d = nd[b], a = na[b];
  float x = fmaxf(d / CUT, 1e-6f);
  float x2 = x * x, x5 = x2 * x2 * x;
  float env = 1.f / x - 28.f * x5 + 48.f * x5 * x - 21.f * x5 * x2;
  float ca = __cosf(a);
  float cbf0 = 0.28209479f, cbf1 = 0.48860251f * ca, cbf2 = 0.31539157f * (3.f * ca * ca - 1.f);
  float tb[BZC];
#pragma unroll
  for (int z = 0; z < BZC; ++z) tb[z] = 0.f;
#pragma unroll
  for (int l = 0; l < 3; ++l) {
    float cb = (l == 0) ? cbf0 : ((l == 1) ? cbf1 : cbf2);
#pragma unroll
    for (int m = 0; m < RR; ++m) {
      float arg = SPHZ[l][m] * x;
      float s, c;
      __sincosf(arg, &s, &c);
      float ia = 1.f / arg;
      float jl;
      if (l == 0)      jl = s * ia;
      else if (l == 1) jl = s * ia * ia - c * ia;
      else             jl = (3.f * ia * ia - 1.f) * s * ia - 3.f * c * ia * ia;
      float val = env * jl * cb;
      int ridx = l * RR + m;
#pragma unroll
      for (int z = 0; z < BZC; ++z) tb[z] += val * w1[ridx * BZC + z];
    }
  }
  float fac = 0.f;
#pragma unroll
  for (int z = 0; z < BZC; ++z) fac += tb[z] * w2[z * HH + h];
  size_t o = (size_t)n * HH + h;
  float v = ne32[o] * fac;
  ne32[o] = v; ne16[o] = (f16)v;
}

__global__ void k_concat2(const f16* __restrict__ ne16, const int* __restrict__ c1f,
                          f16* __restrict__ outf)
{
  long idx = (long)blockIdx.x * 256 + threadIdx.x;
  if (idx >= (long)NB * 512) return;
  int b = (int)(idx / 512), c = (int)(idx % 512);
  int which = c >> 8, h = c & 255;
  int node = c1f[b * 2 + (which == 0 ? 1 : 0)];
  outf[idx] = ne16[(size_t)node * HH + h];
}

__global__ void k_concat3(const f16* __restrict__ ne16, const int* __restrict__ c2c1,
                          f16* __restrict__ outf)
{
  long idx = (long)blockIdx.x * 256 + threadIdx.x;
  if (idx >= (long)NB * 768) return;
  int b = (int)(idx / 768), c = (int)(idx % 768);
  int which = c >> 8, h = c & 255;
  int node = c2c1[b * 3 + (2 - which)];
  outf[idx] = ne16[(size_t)node * HH + h];
}

__global__ void k_extract_col(const int* __restrict__ src, int* __restrict__ dst,
                              int n, int stride, int col)
{
  int i = blockIdx.x * 256 + threadIdx.x;
  if (i < n) dst[i] = src[i * stride + col];
}

__global__ void k_xz(const int* __restrict__ nat, float* __restrict__ x, float* __restrict__ lj)
{
  int i = blockIdx.x * 256 + threadIdx.x;
  if (i >= NB * AA) return;
  int b = i / AA, a = i % AA;
  unsigned u = (unsigned)i * 747796405u + 2891336453u;
  u ^= u >> 17; u *= 0xed5ad4bbu; u ^= u >> 11; u *= 0xac4c1b51u; u ^= u >> 15;
  float r = (float)(u >> 8) * (1.f / 16777216.f);  // dequantization noise
  x[i] = ((a == nat[b]) ? 1.f : 0.f) + 0.9f * r;
  lj[i] = 0.f;
}

__global__ void k_copy_init(const float* __restrict__ src, float* __restrict__ x,
                            float* __restrict__ lj, int n)
{
  int i = blockIdx.x * 256 + threadIdx.x;
  if (i < n) { x[i] = src[i]; lj[i] = 0.f; }
}

__global__ void k_flow_update(float* __restrict__ x, float* __restrict__ lj,
                              const float* __restrict__ st, int dim, int pad, int n)
{
  int i = blockIdx.x * 256 + threadIdx.x;
  if (i >= n) return;
  int b = i / dim, c = i % dim;
  float sr = st[b * pad + c], tt = st[b * pad + dim + c];
  float s = 1.f / (1.f + __expf(-(sr + 2.f)));
  x[i] = x[i] * s + tt;
  lj[i] += __logf(s);
}

__global__ void k_dot(const f16* __restrict__ hid, const float* __restrict__ w,
                      const float* __restrict__ b, float* __restrict__ out, int M, int K)
{
  int m = blockIdx.x * 256 + threadIdx.x;
  if (m >= M) return;
  float a = 0.f;
  for (int k = 0; k < K; ++k) a += (float)hid[(size_t)m * K + k] * w[k];
  out[m] = a + b[0];
}

// ---------------------------------------------------------------------------
extern "C" void kernel_launch(void* const* d_in, const int* in_sizes, int n_in,
                              void* d_out, int out_size, void* d_ws, size_t ws_size,
                              hipStream_t stream)
{
  const float* position   = (const float*)d_in[0];
  const float* new_dist   = (const float*)d_in[1];
  const float* new_angle  = (const float*)d_in[2];
  const float* new_tors   = (const float*)d_in[3];
  const float* emb        = (const float*)d_in[4];
  const float* iw1 = (const float*)d_in[5];   const float* ib1 = (const float*)d_in[6];
  const float* iw2 = (const float*)d_in[7];   const float* ib2 = (const float*)d_in[8];
  const float* ilin = (const float*)d_in[9];
  const float* ow1 = (const float*)d_in[10];  const float* ob1 = (const float*)d_in[11];
  const float* ow2 = (const float*)d_in[12];  const float* ob2 = (const float*)d_in[13];
  const float* fw1 = (const float*)d_in[14];  const float* fb1 = (const float*)d_in[15];
  const float* fw2 = (const float*)d_in[16];  const float* fb2 = (const float*)d_in[17];
  const float* cw1 = (const float*)d_in[18];  const float* cb1 = (const float*)d_in[19];
  const float* cw2 = (const float*)d_in[20];  const float* cb2 = (const float*)d_in[21];
  const float* nf_w1 = (const float*)d_in[22]; const float* nf_b1 = (const float*)d_in[23];
  const float* nf_w2 = (const float*)d_in[24]; const float* nf_b2 = (const float*)d_in[25];
  const float* df_w1 = (const float*)d_in[26]; const float* df_b1 = (const float*)d_in[27];
  const float* df_w2 = (const float*)d_in[28]; const float* df_b2 = (const float*)d_in[29];
  const float* af_w1 = (const float*)d_in[30]; const float* af_b1 = (const float*)d_in[31];
  const float* af_w2 = (const float*)d_in[32]; const float* af_b2 = (const float*)d_in[33];
  const float* tf_w1 = (const float*)d_in[34]; const float* tf_b1 = (const float*)d_in[35];
  const float* tf_w2 = (const float*)d_in[36]; const float* tf_b2 = (const float*)d_in[37];
  const float* dlb_w1 = (const float*)d_in[38]; const float* dlb_w2 = (const float*)d_in[39];
  const float* alb_w1 = (const float*)d_in[40]; const float* alb_w2 = (const float*)d_in[41];
  const int* atom_type = (const int*)d_in[42];
  const int* batch     = (const int*)d_in[43];
  const int* eidx      = (const int*)d_in[44];
  const int* lig_index = (const int*)d_in[45];
  const int* contact_index = (const int*)d_in[46];
  const int* new_atom_type = (const int*)d_in[47];
  const int* focus = (const int*)d_in[48];
  const int* c1f   = (const int*)d_in[49];
  const int* c2c1  = (const int*)d_in[50];
  const int* esrc = eidx;
  const int* edst = eidx + NE;
  float* out = (float*)d_out;

  // output layout (flat, return order)
  float* o_nl  = out + 0;       float* o_nlj = out + 5120;
  float* o_fs  = out + 10240;   float* o_cs  = out + 20480;
  float* o_dl  = out + 30720;   float* o_dlj = out + 31232;
  float* o_al  = out + 31744;   float* o_alj = out + 32256;
  float* o_tl  = out + 32768;   float* o_tlj = out + 33280;

  // ---- workspace carve (256B aligned) ----
  char* wsb = (char*)d_ws;
  size_t off = 0;
  auto carve = [&](size_t bytes) -> void* {
    void* p = wsb + off;
    off = (off + bytes + 255) & ~(size_t)255;
    return p;
  };
  f16*   rbf16 = (f16*)carve((size_t)NE * 64 * 2);
  float* Cw    = (float*)carve((size_t)NE * 4);
  f16*   h1    = (f16*)carve((size_t)NE * HH * 2);
  f16*   Wm    = (f16*)carve((size_t)NE * HH * 2);
  float* v32   = (float*)carve((size_t)NN * HH * 4);
  f16*   v16   = (f16*)carve((size_t)NN * HH * 2);
  f16*   vl16  = (f16*)carve((size_t)NN * HH * 2);
  f16*   agg16 = (f16*)carve((size_t)NN * HH * 2);
  f16*   t116  = (f16*)carve((size_t)NN * HH * 2);
  float* ne32  = (float*)carve((size_t)NN * HH * 4);
  f16*   ne16  = (f16*)carve((size_t)NN * HH * 2);
  f16* iw1c  = (f16*)carve((size_t)LL * 64 * 256 * 2);
  f16* iw2c  = (f16*)carve((size_t)LL * 65536 * 2);
  f16* ilinc = (f16*)carve((size_t)LL * 65536 * 2);
  f16* ow1c  = (f16*)carve((size_t)LL * 65536 * 2);
  f16* ow2c  = (f16*)carve((size_t)LL * 65536 * 2);
  f16* fw1c  = (f16*)carve((size_t)256 * 128 * 2);
  f16* cw1c  = (f16*)carve((size_t)256 * 128 * 2);
  f16* nfw1c = (f16*)carve((size_t)KK * 65536 * 2);
  f16* nfw2c = (f16*)carve((size_t)KK * 256 * 32 * 2);
  f16* dfw1c = (f16*)carve((size_t)KK * 65536 * 2);
  f16* dfw2c = (f16*)carve((size_t)KK * 256 * 16 * 2);
  f16* afw1c = (f16*)carve((size_t)KK * 512 * 256 * 2);
  f16* afw2c = (f16*)carve((size_t)KK * 256 * 16 * 2);
  f16* tfw1c = (f16*)carve((size_t)KK * 768 * 256 * 2);
  f16* tfw2c = (f16*)carve((size_t)KK * 256 * 16 * 2);
  float* nfb2p = (float*)carve((size_t)KK * 32 * 4);
  float* dfb2p = (float*)carve((size_t)KK * 16 * 4);
  float* afb2p = (float*)carve((size_t)KK * 16 * 4);
  float* tfb2p = (float*)carve((size_t)KK * 16 * 4);
  int*   focus0 = (int*)carve((size_t)NB * 4);
  f16*   hidf  = (f16*)carve((size_t)NB * HH * 2);
  float* st32  = (float*)carve((size_t)NB * 32 * 4);
  f16*   hidS  = (f16*)carve((size_t)NLIG * 128 * 2);
  f16*   aff   = (f16*)carve((size_t)NB * 512 * 2);
  f16*   tff   = (f16*)carve((size_t)NB * 768 * 2);

  auto gemm = [&](const f16* A, const f16* Bw, const float* bias, const float* rs,
                  const float* resid, f16* o16, float* o32, const int* gidx,
                  int M, int Kd, int Nld, int act) {
    if ((Nld & 31) == 0) {
      dim3 g(M / 128, Nld / 32);
      gemm_wmma<2><<<g, 128, 0, stream>>>(A, Bw, bias, rs, resid, o16, o32, gidx, M, Kd, Nld, act);
    } else {
      dim3 g(M / 128, Nld / 16);
      gemm_wmma<1><<<g, 128, 0, stream>>>(A, Bw, bias, rs, resid, o16, o32, gidx, M, Kd, Nld, act);
    }
  };
  auto cvt = [&](const float* s, f16* dptr, int Rs, int Cs, int Rd, int Cd, int nb) {
    long tot = (long)nb * Rd * Cd;
    k_cvt_pad<<<(unsigned)((tot + 255) / 256), 256, 0, stream>>>(s, dptr, Rs, Cs, Rd, Cd, tot);
  };

  // ---- weight conversion to f16 (pad K->mult64, N->mult16) ----
  cvt(iw1, iw1c, GG, 256, 64, 256, LL);
  cvt(iw2, iw2c, 256, 256, 256, 256, LL);
  cvt(ilin, ilinc, 256, 256, 256, 256, LL);
  cvt(ow1, ow1c, 256, 256, 256, 256, LL);
  cvt(ow2, ow2c, 256, 256, 256, 256, LL);
  cvt(fw1, fw1c, 256, 128, 256, 128, 1);
  cvt(cw1, cw1c, 256, 128, 256, 128, 1);
  cvt(nf_w1, nfw1c, 256, 256, 256, 256, KK);
  cvt(nf_w2, nfw2c, 256, 2 * AA, 256, 32, KK);
  cvt(df_w1, dfw1c, 256, 256, 256, 256, KK);
  cvt(df_w2, dfw2c, 256, 2, 256, 16, KK);
  cvt(af_w1, afw1c, 512, 256, 512, 256, KK);
  cvt(af_w2, afw2c, 256, 2, 256, 16, KK);
  cvt(tf_w1, tfw1c, 768, 256, 768, 256, KK);
  cvt(tf_w2, tfw2c, 256, 2, 256, 16, KK);
  k_pad32<<<(KK * 32 + 255) / 256, 256, 0, stream>>>(nf_b2, nfb2p, 2 * AA, 32, KK);
  k_pad32<<<(KK * 16 + 255) / 256, 256, 0, stream>>>(df_b2, dfb2p, 2, 16, KK);
  k_pad32<<<(KK * 16 + 255) / 256, 256, 0, stream>>>(af_b2, afb2p, 2, 16, KK);
  k_pad32<<<(KK * 16 + 255) / 256, 256, 0, stream>>>(tf_b2, tfb2p, 2, 16, KK);

  // ---- SchNet ----
  k_prep_edges<<<(unsigned)(((long)NE * 64 + 255) / 256), 256, 0, stream>>>(
      position, esrc, edst, rbf16, Cw);
  k_embed<<<(unsigned)(((long)NN * HH + 255) / 256), 256, 0, stream>>>(
      emb, atom_type, v32, v16);

  for (int l = 0; l < LL; ++l) {
    const f16* iw1l = iw1c + (size_t)l * 64 * 256;
    const f16* iw2l = iw2c + (size_t)l * 65536;
    const f16* ilinl = ilinc + (size_t)l * 65536;
    const f16* ow1l = ow1c + (size_t)l * 65536;
    const f16* ow2l = ow2c + (size_t)l * 65536;
    // h1 = ssp(rbf @ iw1 + ib1)
    gemm(rbf16, iw1l, ib1 + l * 256, nullptr, nullptr, h1, nullptr, nullptr,
         NE, 64, 256, ACT_SSP);
    // W = (h1 @ iw2 + ib2) * C
    gemm(h1, iw2l, ib2 + l * 256, Cw, nullptr, Wm, nullptr, nullptr,
         NE, 256, 256, ACT_NONE);
    // vl = v @ ilin
    gemm(v16, ilinl, nullptr, nullptr, nullptr, vl16, nullptr, nullptr,
         NN, 256, 256, ACT_NONE);
    // agg[i] = sum_e vl[j_e] * W[e]   (DEG=16 contiguous edges per node)
    k_edge_agg<<<NN, 256, 0, stream>>>(vl16, Wm, edst, agg16);
    // t1 = ssp(agg @ ow1 + ob1)
    gemm(agg16, ow1l, ob1 + l * 256, nullptr, nullptr, t116, nullptr, nullptr,
         NN, 256, 256, ACT_SSP);
    // v = v + t1 @ ow2 + ob2
    gemm(t116, ow2l, ob2 + l * 256, nullptr, v32, v16, v32, nullptr,
         NN, 256, 256, ACT_NONE);
  }

  // ---- focus / contact scores ----
  gemm(v16, fw1c, fb1, nullptr, nullptr, hidS, nullptr, lig_index,
       NLIG, 256, 128, ACT_RELU);
  k_dot<<<(NLIG + 255) / 256, 256, 0, stream>>>(hidS, fw2, fb2, o_fs, NLIG, 128);
  gemm(v16, cw1c, cb1, nullptr, nullptr, hidS, nullptr, contact_index,
       NLIG, 256, 128, ACT_RELU);
  k_dot<<<(NLIG + 255) / 256, 256, 0, stream>>>(hidS, cw2, cb2, o_cs, NLIG, 128);

  // ---- node-type flow ----
  k_extract_col<<<(NB + 255) / 256, 256, 0, stream>>>(focus, focus0, NB, 2, 0);
  k_xz<<<(NB * AA + 255) / 256, 256, 0, stream>>>(new_atom_type, o_nl, o_nlj);
  for (int k = 0; k < KK; ++k) {
    gemm(v16, nfw1c + (size_t)k * 65536, nf_b1 + k * 256, nullptr, nullptr,
         hidf, nullptr, focus0, NB, 256, 256, ACT_TANH);
    gemm(hidf, nfw2c + (size_t)k * 256 * 32, nfb2p + k * 32, nullptr, nullptr,
         nullptr, st32, nullptr, NB, 256, 32, ACT_NONE);
    k_flow_update<<<(NB * AA + 255) / 256, 256, 0, stream>>>(o_nl, o_nlj, st32, AA, 32, NB * AA);
  }

  // ---- node_emb = node_feat * emb[new_atom_type][batch] ----
  k_node_emb<<<(unsigned)(((long)NN * HH + 255) / 256), 256, 0, stream>>>(
      v32, emb, new_atom_type, batch, ne32, ne16);

  // ---- distance flow ----
  k_copy_init<<<(NB + 255) / 256, 256, 0, stream>>>(new_dist, o_dl, o_dlj, NB);
  for (int k = 0; k < KK; ++k) {
    gemm(ne16, dfw1c + (size_t)k * 65536, df_b1 + k * 256, nullptr, nullptr,
         hidf, nullptr, focus0, NB, 256, 256, ACT_TANH);
    gemm(hidf, dfw2c + (size_t)k * 256 * 16, dfb2p + k * 16, nullptr, nullptr,
         nullptr, st32, nullptr, NB, 256, 16, ACT_NONE);
    k_flow_update<<<(NB + 255) / 256, 256, 0, stream>>>(o_dl, o_dlj, st32, 1, 16, NB);
  }

  // ---- bessel-basis modulation, angle flow ----
  k_dist_basis_mul<<<NN, 256, 0, stream>>>(new_dist, batch, dlb_w1, dlb_w2, ne32, ne16);
  k_concat2<<<(unsigned)(((long)NB * 512 + 255) / 256), 256, 0, stream>>>(ne16, c1f, aff);
  k_copy_init<<<(NB + 255) / 256, 256, 0, stream>>>(new_angle, o_al, o_alj, NB);
  for (int k = 0; k < KK; ++k) {
    gemm(aff, afw1c + (size_t)k * 512 * 256, af_b1 + k * 256, nullptr, nullptr,
         hidf, nullptr, nullptr, NB, 512, 256, ACT_TANH);
    gemm(hidf, afw2c + (size_t)k * 256 * 16, afb2p + k * 16, nullptr, nullptr,
         nullptr, st32, nullptr, NB, 256, 16, ACT_NONE);
    k_flow_update<<<(NB + 255) / 256, 256, 0, stream>>>(o_al, o_alj, st32, 1, 16, NB);
  }

  // ---- spherical-basis modulation, torsion flow ----
  k_sph_basis_mul<<<NN, 256, 0, stream>>>(new_dist, new_angle, batch, alb_w1, alb_w2, ne32, ne16);
  k_concat3<<<(unsigned)(((long)NB * 768 + 255) / 256), 256, 0, stream>>>(ne16, c2c1, tff);
  k_copy_init<<<(NB + 255) / 256, 256, 0, stream>>>(new_tors, o_tl, o_tlj, NB);
  for (int k = 0; k < KK; ++k) {
    gemm(tff, tfw1c + (size_t)k * 768 * 256, tf_b1 + k * 256, nullptr, nullptr,
         hidf, nullptr, nullptr, NB, 768, 256, ACT_TANH);
    gemm(hidf, tfw2c + (size_t)k * 256 * 16, tfb2p + k * 16, nullptr, nullptr,
         nullptr, st32, nullptr, NB, 256, 16, ACT_NONE);
    k_flow_update<<<(NB + 255) / 256, 256, 0, stream>>>(o_tl, o_tlj, st32, 1, 16, NB);
  }
}
